// DEGT_81724637708946
// MI455X (gfx1250) — matrix operand order repaired and checked by reading, not compiled, not run on hardware
//
#include <hip/hip_runtime.h>
#include <hip/hip_bf16.h>
#include <math.h>

// ---------------- sizes (match reference) ----------------
#define NN   50000
#define NE   500000
#define FIN  128
#define HC   256
#define NH   16
#define HD   16
#define OC   32

typedef __attribute__((ext_vector_type(16))) __bf16 v16bf;
typedef __attribute__((ext_vector_type(8)))  float  v8f;
typedef __attribute__((ext_vector_type(4)))  int    v4i;

union FragB { uint4 u[2]; v16bf v; };

#if __has_builtin(__builtin_amdgcn_global_load_async_to_lds_b128)
#define USE_ASYNC_LDS 1
typedef __attribute__((address_space(1))) v4i gv4i;   // global int4
typedef __attribute__((address_space(3))) v4i lv4i;   // LDS int4
#else
#define USE_ASYNC_LDS 0
#endif

__device__ __forceinline__ float gelu_exact(float x) {
    return 0.5f * x * (1.0f + erff(x * 0.70710678118654752440f));
}

__device__ __forceinline__ unsigned short f2bf_rne(float f) {
    unsigned u = __float_as_uint(f);
    unsigned r = (u + 0x7FFFu + ((u >> 16) & 1u)) >> 16;
    return (unsigned short)r;
}

// ---------------- weight packing into WMMA B-fragment order ----------------
// B fragment (kb,nb): 32 lanes x 8 dwords. lane<16 -> K rows kb*32+0..15,
// lane>=16 -> K rows kb*32+16..31; dword v holds K=2v (lo16) and K=2v+1 (hi16);
// column N = nb*16 + (lane&15).   W is [K, Nc] row-major f32.
__global__ void pack_weight_kernel(const float* __restrict__ W,
                                   unsigned* __restrict__ out, int K, int Nc) {
    int tid = blockIdx.x * 256 + threadIdx.x;
    int total = (K * Nc) >> 1;
    if (tid >= total) return;
    int v    = tid & 7;
    int lane = (tid >> 3) & 31;
    int frag = tid >> 8;
    int NB = Nc >> 4;
    int nb = frag % NB;
    int kb = frag / NB;
    int n  = nb * 16 + (lane & 15);
    int k0 = kb * 32 + ((lane >> 4) << 4) + 2 * v;
    unsigned lo = f2bf_rne(W[(size_t)k0 * Nc + n]);
    unsigned hi = f2bf_rne(W[(size_t)(k0 + 1) * Nc + n]);
    out[tid] = (hi << 16) | lo;
}

// ---------------- WMMA GEMM: C = epi(A @ W + bias) ----------------
// A: [M,K] f32 row-major.  Bp: packed bf16 fragments.  C: [M,Nc] f32.
// Block = 128 threads (4 waves). Wave w computes rows [blkM + 16w, +16),
// cols [nBase, nBase + WN*16). Whole B panel (K x WN*16) staged in LDS once
// per block (async-to-LDS when available); K-loop software-pipelined x2.
// EPI: 0 = bias only, 1 = bias + exact GELU.

__device__ __forceinline__ void load_a_frag(const float* __restrict__ ap,
                                            int kOff0, int kOff1, float4* r) {
    r[0] = *(const float4*)(ap + kOff0);
    r[1] = *(const float4*)(ap + kOff0 + 4);
    r[2] = *(const float4*)(ap + kOff1);
    r[3] = *(const float4*)(ap + kOff1 + 4);
}

__device__ __forceinline__ v16bf cvt_a_frag(const float4* r) {
    v16bf av;
    av[0]  = (__bf16)r[0].x; av[1]  = (__bf16)r[0].y; av[2]  = (__bf16)r[0].z; av[3]  = (__bf16)r[0].w;
    av[4]  = (__bf16)r[1].x; av[5]  = (__bf16)r[1].y; av[6]  = (__bf16)r[1].z; av[7]  = (__bf16)r[1].w;
    av[8]  = (__bf16)r[2].x; av[9]  = (__bf16)r[2].y; av[10] = (__bf16)r[2].z; av[11] = (__bf16)r[2].w;
    av[12] = (__bf16)r[3].x; av[13] = (__bf16)r[3].y; av[14] = (__bf16)r[3].z; av[15] = (__bf16)r[3].w;
    return av;
}

template <int WN, int EPI>
__global__ __launch_bounds__(128) void wmma_gemm_kernel(
    const float* __restrict__ A, const unsigned* __restrict__ Bp,
    const float* __restrict__ bias, float* __restrict__ C,
    int M, int K, int Nc) {
    // B panel: KB (<=16) slices x WN frags x 256 dwords
    __shared__ unsigned ldsB[16 * WN * 256];

    const int lane  = threadIdx.x & 31;
    const int wave  = threadIdx.x >> 5;
    const int mBase = (blockIdx.y * 4 + wave) * 16;
    const int nBase  = blockIdx.x * (WN * 16);
    const int laneLo = lane & 15;
    const int hi     = lane >> 4;
    const int kOff0 = hi * 16;             // A layout: lane<16 -> K {0..15}
    const int kOff1 = kOff0 + 8;           //           lane>=16-> K {16..31}
    const int NB     = Nc >> 4;
    const int nbBase = nBase >> 4;
    const int KB = K >> 5;

    // ---- cooperative stage of B panel into LDS ----
    {
        const int t = threadIdx.x;         // 0..127
        for (int kb = 0; kb < KB; ++kb) {
            const uint4* src = (const uint4*)(Bp + ((size_t)(kb * NB + nbBase)) * 256);
            unsigned* dstBase = &ldsB[kb * WN * 256];
#pragma unroll
            for (int i = t; i < WN * 64; i += 128) {
#if USE_ASYNC_LDS
                __builtin_amdgcn_global_load_async_to_lds_b128(
                    (gv4i*)(src + i), (lv4i*)(dstBase + i * 4), 0, 0);
#else
                uint4 d = src[i];
                *(uint4*)(dstBase + i * 4) = d;
#endif
            }
        }
#if USE_ASYNC_LDS
#if __has_builtin(__builtin_amdgcn_s_wait_asynccnt)
        __builtin_amdgcn_s_wait_asynccnt(0);
#else
        asm volatile("s_wait_asynccnt 0x0" ::: "memory");
#endif
#endif
        __syncthreads();
    }

    if (mBase < M) {                       // wave-uniform: EXEC stays all-ones
        const size_t rowOff = (size_t)(mBase + laneLo) * K;

        v8f acc[WN];
#pragma unroll
        for (int j = 0; j < WN; ++j)
#pragma unroll
            for (int r = 0; r < 8; ++r) acc[j][r] = 0.0f;

        float4 aA[4], aB[4];
        load_a_frag(A + rowOff, kOff0, kOff1, aA);

        // KB is always even (4, 8 or 16): pipeline x2, next-A issued before WMMAs
        for (int kb = 0; kb < KB; kb += 2) {
            load_a_frag(A + rowOff + (size_t)(kb + 1) * 32, kOff0, kOff1, aB);
            {
                v16bf av = cvt_a_frag(aA);
#pragma unroll
                for (int j = 0; j < WN; ++j) {
                    const uint4* bp = (const uint4*)&ldsB[(kb * WN + j) * 256 + lane * 8];
                    FragB bf_;
                    bf_.u[0] = bp[0];
                    bf_.u[1] = bp[1];
                    acc[j] = __builtin_amdgcn_wmma_f32_16x16x32_bf16(
                        false, av, false, bf_.v, (short)0, acc[j], false, false);
                }
            }
            if (kb + 2 < KB)
                load_a_frag(A + rowOff + (size_t)(kb + 2) * 32, kOff0, kOff1, aA);
            {
                v16bf av = cvt_a_frag(aB);
#pragma unroll
                for (int j = 0; j < WN; ++j) {
                    const uint4* bp = (const uint4*)&ldsB[((kb + 1) * WN + j) * 256 + lane * 8];
                    FragB bf_;
                    bf_.u[0] = bp[0];
                    bf_.u[1] = bp[1];
                    acc[j] = __builtin_amdgcn_wmma_f32_16x16x32_bf16(
                        false, av, false, bf_.v, (short)0, acc[j], false, false);
                }
            }
        }

        // epilogue: D layout: VGPR r -> row mBase + hi*8 + r, col nBase + j*16 + laneLo
#pragma unroll
        for (int j = 0; j < WN; ++j) {
            const int n  = nBase + j * 16 + laneLo;
            const float bs = bias[n];
#pragma unroll
            for (int r = 0; r < 8; ++r) {
                float val = acc[j][r] + bs;
                if (EPI == 1) val = gelu_exact(val);
                C[(size_t)(mBase + hi * 8 + r) * Nc + n] = val;
            }
        }
    }
}

// ---------------- edge weight: ew = (edge_emb + dire_emb) * cos(t * freq) ----
__global__ void edge_ew_kernel(const float* __restrict__ et,
                               const int* __restrict__ eattr,
                               const int* __restrict__ ed,
                               const float* __restrict__ edge_emb,
                               const float* __restrict__ dire_emb,
                               float* __restrict__ ew, int E) {
    int i = blockIdx.x * 256 + threadIdx.x;
    if (i >= E * NH) return;
    int e = i >> 4, h = i & 15;
    float fr = powf(10.0f, -0.6f * (float)h);     // 1/10^(9h/15)
    float te = cosf(et[e] * fr);
    ew[i] = (edge_emb[eattr[e] * NH + h] + dire_emb[ed[e] * NH + h]) * te;
}

// ---------------- SDDMM on edges + sigmoid gate -----------------------------
// attn[e,h] = sigmoid( 0.25 * <q[dst[e],:,h], k[src[e],:,h]> ) * ew[e,h]
__global__ void sddmm_kernel(const float* __restrict__ q,
                             const float* __restrict__ k,
                             const int* __restrict__ dst,
                             const int* __restrict__ src,
                             const float* __restrict__ ew,
                             float* __restrict__ attn, int E) {
    int i = blockIdx.x * 256 + threadIdx.x;
    if (i >= E * NH) return;
    int e = i >> 4, h = i & 15;
    const float* qp = q + (size_t)dst[e] * HC + h;
    const float* kp = k + (size_t)src[e] * HC + h;
    float s = 0.0f;
#pragma unroll
    for (int d = 0; d < HD; ++d) s += qp[d * NH] * kp[d * NH];
    s *= 0.25f;                                   // HD^-0.5
    attn[i] = ew[i] / (1.0f + __expf(-s));
}

// ---------------- SpMM scatter: agg[dst] += attn * v[src] -------------------
__global__ void spmm_scatter_kernel(const float* __restrict__ attn,
                                    const float* __restrict__ v,
                                    const int* __restrict__ dst,
                                    const int* __restrict__ src,
                                    float* __restrict__ agg) {
    int e = blockIdx.x;
    int c = threadIdx.x;                          // 0..255, h = c&15
    int s = src[e], d = dst[e];
    float m = attn[(size_t)e * NH + (c & 15)] * v[(size_t)s * HC + c];
    unsafeAtomicAdd(agg + (size_t)d * HC + c, m); // global_atomic_add_f32
}

// ---------------- residual + BN batch statistics ----------------------------
__global__ void residual_stats_kernel(const float* __restrict__ h,
                                      float* __restrict__ agg,   // in: agg, out: z
                                      float* __restrict__ sums,  // [512]
                                      int Nn) {
    int c = threadIdx.x;
    int r0 = blockIdx.x * 64;
    int r1 = r0 + 64; if (r1 > Nn) r1 = Nn;
    float s = 0.0f, s2 = 0.0f;
    for (int r = r0; r < r1; ++r) {
        size_t idx = (size_t)r * HC + c;
        float z = h[idx] + agg[idx];
        agg[idx] = z;
        s += z; s2 += z * z;
    }
    unsafeAtomicAdd(&sums[c], s);
    unsafeAtomicAdd(&sums[HC + c], s2);
}

__global__ void bn_finalize_kernel(const float* __restrict__ sums,
                                   const float* __restrict__ g,
                                   const float* __restrict__ b,
                                   float* __restrict__ ab, float invN) {
    int c = threadIdx.x;
    float mean = sums[c] * invN;
    float var  = sums[HC + c] * invN - mean * mean;
    float a = g[c] * rsqrtf(var + 1e-5f);
    ab[c] = a;
    ab[HC + c] = b[c] - mean * a;
}

__global__ void bn_apply_kernel(const float* __restrict__ z,
                                const float* __restrict__ ab,
                                float* __restrict__ out, long total) {
    long i = (long)blockIdx.x * 256 + threadIdx.x;
    if (i >= total) return;
    int c = (int)(i & (HC - 1));
    out[i] = ab[c] * z[i] + ab[HC + c];
}

__global__ void fill_zero_kernel(float* __restrict__ p, long n) {
    long i = (long)blockIdx.x * 256 + threadIdx.x;
    if (i < n) p[i] = 0.0f;
}

// ---------------- host orchestration ----------------------------------------
extern "C" void kernel_launch(void* const* d_in, const int* in_sizes, int n_in,
                              void* d_out, int out_size, void* d_ws, size_t ws_size,
                              hipStream_t stream) {
    const float* x      = (const float*)d_in[0];
    const int*   eidx   = (const int*)d_in[1];
    const int*   eattr  = (const int*)d_in[2];
    const float* et     = (const float*)d_in[3];
    const int*   ed     = (const int*)d_in[4];
    const int* dst = eidx;           // edge_index[0]
    const int* src = eidx + NE;      // edge_index[1]

    const float* ne_w1 = (const float*)d_in[5];
    const float* ne_b1 = (const float*)d_in[6];
    const float* ne_w2 = (const float*)d_in[7];
    const float* ne_b2 = (const float*)d_in[8];
    const float* edge_emb = (const float*)d_in[9];
    const float* dire_emb = (const float*)d_in[10];
    // per layer base: 11 + 12*l : qw qb kw kb vw vb bn_g bn_b f1w f1b f2w f2b
    const float* agg_w  = (const float*)d_in[35];
    const float* agg_b  = (const float*)d_in[36];
    const float* pred_w = (const float*)d_in[37];
    const float* pred_b = (const float*)d_in[38];

    // ---- workspace bump allocator ----
    char* wsp = (char*)d_ws;
    size_t off = 0;
    auto alloc = [&](size_t bytes) -> void* {
        void* p = wsp + off;
        off += (bytes + 255) & ~(size_t)255;
        return p;
    };
    const size_t NHCf = (size_t)NN * HC;
    float* bufH1  = (float*)alloc(NHCf * 4);
    float* bufH2  = (float*)alloc(NHCf * 4);
    float* bufQK  = (float*)alloc(NHCf * 8);      // q|k contiguous; reused as FFN mid [N,512]
    float* bufQ   = bufQK;
    float* bufK   = bufQK + NHCf;
    float* bufV   = (float*)alloc(NHCf * 4);
    float* bufAgg = (float*)alloc(NHCf * 4);
    float* bufEw  = (float*)alloc((size_t)NE * NH * 4);
    float* bufAtt = (float*)alloc((size_t)NE * NH * 4);
    float* bufStat= (float*)alloc(1024 * 4);      // sums[512] then ab[512]
    float* sums   = bufStat;
    float* ab     = bufStat + 512;

    // ---- pack weights to bf16 WMMA fragments ----
    auto pack_off = [&](const float* W, int K, int Nc, unsigned** cur) -> unsigned* {
        unsigned* o = *cur;
        int total = (K * Nc) >> 1;
        pack_weight_kernel<<<(total + 255) / 256, 256, 0, stream>>>(W, o, K, Nc);
        *cur += total;
        return o;
    };
    unsigned* wpack = (unsigned*)alloc(600000 * 4);
    unsigned* cur = wpack;
    unsigned* p_ne1 = pack_off(ne_w1, FIN, HC, &cur);
    unsigned* p_ne2 = pack_off(ne_w2, HC, HC, &cur);
    unsigned* p_q[2]; unsigned* p_k[2]; unsigned* p_v[2];
    unsigned* p_f1[2]; unsigned* p_f2[2];
    for (int l = 0; l < 2; ++l) {
        int b = 11 + 12 * l;
        p_q[l]  = pack_off((const float*)d_in[b + 0], HC, HC, &cur);
        p_k[l]  = pack_off((const float*)d_in[b + 2], HC, HC, &cur);
        p_v[l]  = pack_off((const float*)d_in[b + 4], HC, HC, &cur);
        p_f1[l] = pack_off((const float*)d_in[b + 8], HC, 2 * HC, &cur);
        p_f2[l] = pack_off((const float*)d_in[b + 10], 2 * HC, HC, &cur);
    }
    unsigned* p_agg  = pack_off(agg_w, HC, HC, &cur);
    unsigned* p_pred = pack_off(pred_w, HC, OC, &cur);

    const int gridY = (NN / 16 + 3) / 4;          // 782 (guarded tail waves)
    auto gemm = [&](const float* A, const unsigned* Bp, const float* bias,
                    float* C, int M, int K, int Nc, int epi) {
        dim3 g(Nc / 64, gridY);
        if (epi) wmma_gemm_kernel<4, 1><<<g, 128, 0, stream>>>(A, Bp, bias, C, M, K, Nc);
        else     wmma_gemm_kernel<4, 0><<<g, 128, 0, stream>>>(A, Bp, bias, C, M, K, Nc);
    };

    // ---- edge gate weights (layer-invariant) ----
    edge_ew_kernel<<<(NE * NH + 255) / 256, 256, 0, stream>>>(
        et, eattr, ed, edge_emb, dire_emb, bufEw, NE);

    // ---- node embed MLP ----
    gemm(x, p_ne1, ne_b1, bufH1, NN, FIN, HC, 1);
    gemm(bufH1, p_ne2, ne_b2, bufH2, NN, HC, HC, 1);

    // ---- transformer layers ----
    for (int l = 0; l < 2; ++l) {
        int b = 11 + 12 * l;
        const float* qb   = (const float*)d_in[b + 1];
        const float* kb   = (const float*)d_in[b + 3];
        const float* vb   = (const float*)d_in[b + 5];
        const float* bn_g = (const float*)d_in[b + 6];
        const float* bn_b = (const float*)d_in[b + 7];
        const float* f1b  = (const float*)d_in[b + 9];
        const float* f2b  = (const float*)d_in[b + 11];

        gemm(bufH2, p_q[l], qb, bufQ, NN, HC, HC, 0);
        gemm(bufH2, p_k[l], kb, bufK, NN, HC, HC, 0);
        gemm(bufH2, p_v[l], vb, bufV, NN, HC, HC, 0);

        sddmm_kernel<<<(NE * NH + 255) / 256, 256, 0, stream>>>(
            bufQ, bufK, dst, src, bufEw, bufAtt, NE);

        fill_zero_kernel<<<(int)((NHCf + 255) / 256), 256, 0, stream>>>(bufAgg, (long)NHCf);
        spmm_scatter_kernel<<<NE, 256, 0, stream>>>(bufAtt, bufV, dst, src, bufAgg);

        fill_zero_kernel<<<2, 256, 0, stream>>>(sums, 512);
        residual_stats_kernel<<<(NN + 63) / 64, 256, 0, stream>>>(bufH2, bufAgg, sums, NN);
        bn_finalize_kernel<<<1, 256, 0, stream>>>(sums, bn_g, bn_b, ab, 1.0f / (float)NN);
        bn_apply_kernel<<<(int)((NHCf + 255) / 256), 256, 0, stream>>>(
            bufAgg, ab, bufH1, (long)NHCf);

        gemm(bufH1, p_f1[l], f1b, bufQK, NN, HC, 2 * HC, 1);   // mid = gelu(hn@f1+b)
        gemm(bufQK, p_f2[l], f2b, bufH2, NN, 2 * HC, HC, 0);   // h = mid@f2+b
    }

    // ---- heads ----
    gemm(bufH2, p_agg, agg_b, bufH1, NN, HC, HC, 0);
    {   // pred: Nc = 32 -> WN=2 variant
        dim3 g(OC / 32, gridY);
        wmma_gemm_kernel<2, 0><<<g, 128, 0, stream>>>(
            bufH1, p_pred, pred_b, (float*)d_out, NN, HC, OC);
    }
}
